// MultiHeadAttentionSelf_65481071401597
// MI455X (gfx1250) — matrix operand order repaired and checked by reading, not compiled, not run on hardware
//
#include <hip/hip_runtime.h>
#include <hip/hip_bf16.h>
#include <math.h>

// Problem constants (fixed by the reference)
#define N_TOK  2048          // B*S
#define DMODEL 1024
#define NHEAD  16
#define HD     64
// softmax in base-2: scale = (1/sqrt(1024)) * log2(e)
#define SCALE2 0.04508439f

typedef __attribute__((ext_vector_type(16))) __bf16 v16bf;
typedef __attribute__((ext_vector_type(8)))  float  v8f;

union Frag { uint4 u[2]; v16bf v; };

__device__ __forceinline__ v8f zero8() {
  v8f z = {0.f,0.f,0.f,0.f,0.f,0.f,0.f,0.f};
  return z;
}

// single-instruction base-2 exponential (v_exp_f32)
__device__ __forceinline__ float exp2_fast(float x) {
  return __builtin_amdgcn_exp2f(x);
}

// A-fragment (16x32 bf16, row-major source, ld elements).
// ISA 7.12.2: lane L holds row m=L&15; K runs are [hi*8 .. +7] and [16+hi*8 .. +7].
__device__ __forceinline__ v16bf load_a(const __bf16* src, int row0, int ld,
                                        int k0, int lane) {
  int m = lane & 15, hi = lane >> 4;
  const __bf16* p = src + (size_t)(row0 + m) * ld + k0 + hi * 8;
  Frag f;
  f.u[0] = *(const uint4*)(p);
  f.u[1] = *(const uint4*)(p + 16);
  return f.v;
}

// B-fragment (32x16 bf16). Source is "B transposed" row-major: src[col][k].
// Lane L holds column c=L&15; K runs are [hi*8 .. +7] and [16+hi*8 .. +7].
__device__ __forceinline__ v16bf load_b(const __bf16* src, int col0, int ld,
                                        int k0, int lane) {
  int c = lane & 15, hi = lane >> 4;
  const __bf16* p = src + (size_t)(col0 + c) * ld + k0 + hi * 8;
  Frag f;
  f.u[0] = *(const uint4*)(p);
  f.u[1] = *(const uint4*)(p + 16);
  return f.v;
}

__device__ __forceinline__ v8f bmma(v16bf a, v16bf b, v8f c) {
  return __builtin_amdgcn_wmma_f32_16x16x32_bf16(false, a, false, b, (short)0, c,
                                                 false, false);
}

// Async global -> LDS, 16 bytes per lane (gfx1250 GLOBAL_LOAD_ASYNC_TO_LDS_B128).
__device__ __forceinline__ void async_g2l_b128(uint32_t lds_off, const void* gaddr) {
  asm volatile("global_load_async_to_lds_b128 %0, %1, off"
               :: "v"(lds_off), "v"((uint64_t)(uintptr_t)gaddr)
               : "memory");
}

// ---------------- conversion kernels ----------------

__global__ void __launch_bounds__(256) cvt_bf16(const float* __restrict__ in,
                                                __bf16* __restrict__ out, int n) {
  int i = blockIdx.x * blockDim.x + threadIdx.x;
  if (i < n) out[i] = (__bf16)in[i];
}

// wq/wk [H][HD][HD] -> transposed bf16 [H][HD(e)][HD(d)] so B-fragments are contiguous.
__global__ void __launch_bounds__(256) cvt_wt(const float* __restrict__ wq,
                                              const float* __restrict__ wk,
                                              __bf16* __restrict__ wqT,
                                              __bf16* __restrict__ wkT) {
  int i = blockIdx.x * blockDim.x + threadIdx.x;   // H*HD*HD = 65536
  int h = i >> 12, d = (i >> 6) & 63, e = i & 63;
  int o = (h << 12) + (e << 6) + d;
  wqT[o] = (__bf16)wq[i];
  wkT[o] = (__bf16)wk[i];
}

// ---------------- proj GEMM: flat = x @ W^T + b ----------------
// Grid (32,8): 64-row x 128-col tiles. Block = 4 waves (2x2), each wave 32x64 (2x4 WMMA).
__global__ void __launch_bounds__(128) proj_gemm(const __bf16* __restrict__ xb,
                                                 const __bf16* __restrict__ wb,
                                                 const float*  __restrict__ bias,
                                                 __bf16* __restrict__ flat,
                                                 __bf16* __restrict__ flatT) {
  int lane = threadIdx.x & 31;
  int wave = threadIdx.x >> 5;
  int wm = wave >> 1, wn = wave & 1;
  int row0 = blockIdx.x * 64 + wm * 32;        // token rows
  int col0 = blockIdx.y * 128 + wn * 64;       // embed cols

  v8f acc[2][4];
#pragma unroll
  for (int i = 0; i < 2; ++i)
#pragma unroll
    for (int j = 0; j < 4; ++j) acc[i][j] = zero8();

  for (int k0 = 0; k0 < DMODEL; k0 += 32) {
    v16bf a0 = load_a(xb, row0,      DMODEL, k0, lane);
    v16bf a1 = load_a(xb, row0 + 16, DMODEL, k0, lane);
#pragma unroll
    for (int t = 0; t < 4; ++t) {
      v16bf b = load_b(wb, col0 + t * 16, DMODEL, k0, lane);
      acc[0][t] = bmma(a0, b, acc[0][t]);
      acc[1][t] = bmma(a1, b, acc[1][t]);
    }
  }

  int c = lane & 15, hi = lane >> 4;
#pragma unroll
  for (int mi = 0; mi < 2; ++mi)
#pragma unroll
    for (int ni = 0; ni < 4; ++ni) {
      int e  = col0 + ni * 16 + c;             // global embed index
      int h  = e >> 6;
      int d  = e & 63;
      float bv = bias[e];
#pragma unroll
      for (int r = 0; r < 8; ++r) {
        int n = row0 + mi * 16 + hi * 8 + r;
        float val = acc[mi][ni][r] + bv;
        flat [((size_t)h * N_TOK + n) * HD + d] = (__bf16)val;
        flatT[((size_t)h * HD + d) * N_TOK + n] = (__bf16)val;
      }
    }
}

// ---------------- per-head Q/K GEMM: q = flat@wq, k = flat@wk ----------------
__global__ void __launch_bounds__(128) qk_gemm(const __bf16* __restrict__ flat,
                                               const __bf16* __restrict__ wqT,
                                               const __bf16* __restrict__ wkT,
                                               __bf16* __restrict__ qb,
                                               __bf16* __restrict__ kb) {
  int lane = threadIdx.x & 31;
  int wave = threadIdx.x >> 5;
  int h    = blockIdx.y;
  int row0 = blockIdx.x * 64 + wave * 16;

  const __bf16* fh  = flat + (size_t)h * N_TOK * HD;
  const __bf16* wqh = wqT  + (size_t)h * HD * HD;
  const __bf16* wkh = wkT  + (size_t)h * HD * HD;

  v8f aq[4], ak[4];
#pragma unroll
  for (int t = 0; t < 4; ++t) { aq[t] = zero8(); ak[t] = zero8(); }

#pragma unroll
  for (int k0 = 0; k0 < HD; k0 += 32) {
    v16bf a = load_a(fh, row0, HD, k0, lane);
#pragma unroll
    for (int t = 0; t < 4; ++t) {
      v16bf bq = load_b(wqh, t * 16, HD, k0, lane);
      v16bf bk = load_b(wkh, t * 16, HD, k0, lane);
      aq[t] = bmma(a, bq, aq[t]);
      ak[t] = bmma(a, bk, ak[t]);
    }
  }

  int c = lane & 15, hi = lane >> 4;
#pragma unroll
  for (int t = 0; t < 4; ++t) {
    int d = t * 16 + c;
#pragma unroll
    for (int r = 0; r < 8; ++r) {
      int n = row0 + hi * 8 + r;
      size_t o = ((size_t)h * N_TOK + n) * HD + d;
      qb[o] = (__bf16)aq[t][r];
      kb[o] = (__bf16)ak[t][r];
    }
  }
}

// ---------------- flash attention: softmax(q k^T / 32) @ V ----------------
// Grid (32, 16=head). Block = 4 waves / 64 query rows. Keys staged into LDS in
// 32-key tiles with async global->LDS (double-buffered, ASYNCcnt-tracked).
__global__ void __launch_bounds__(128) attn_kernel(const __bf16* __restrict__ qb,
                                                   const __bf16* __restrict__ kb,
                                                   const __bf16* __restrict__ vT,
                                                   float* __restrict__ out) {
  __shared__ __bf16 kbuf[2][32][64];           // 2 x 4KB: K tile [key][e]
  __shared__ __bf16 vbuf[2][64][32];           // 2 x 4KB: V tile [d][key]
  __shared__ __bf16 pbuf[4][16][32];           // per-wave P staging

  int tid  = threadIdx.x;
  int lane = tid & 31;
  int wave = tid >> 5;
  int h    = blockIdx.y;
  int r0   = blockIdx.x * 64 + wave * 16;
  int c  = lane & 15, hi = lane >> 4;

  const __bf16* qh  = qb + (size_t)h * N_TOK * HD;   // [N][64]
  const __bf16* kh  = kb + (size_t)h * N_TOK * HD;   // [N][64]
  const __bf16* vTh = vT + (size_t)h * HD * N_TOK;   // [64][N]

  // Q fragments for this wave's 16 rows (both 32-wide K chunks of the head dim).
  v16bf qa0 = load_a(qh, r0, HD, 0,  lane);
  v16bf qa1 = load_a(qh, r0, HD, 32, lane);

  float mrow[8], lrow[8];
#pragma unroll
  for (int r = 0; r < 8; ++r) { mrow[r] = -INFINITY; lrow[r] = 0.f; }
  v8f acc[4];
#pragma unroll
  for (int t = 0; t < 4; ++t) acc[t] = zero8();

  __bf16* pb = &pbuf[wave][0][0];

  // stage tile j (32 keys) into LDS buffer `buf`: 4 async b128 ops / thread total
  auto stage = [&](int j, int buf) {
    // K tile: 32 rows x 128B, contiguous 4KB region
    const char* kg = (const char*)(kh + (size_t)j * HD);
    uint32_t kl = (uint32_t)(uintptr_t)&kbuf[buf][0][0];
#pragma unroll
    for (int p = 0; p < 2; ++p) {
      uint32_t off = (uint32_t)(tid + 128 * p) * 16u;
      async_g2l_b128(kl + off, kg + off);
    }
    // V tile: 64 rows (d) x 64B segments of vT rows
    uint32_t vl = (uint32_t)(uintptr_t)&vbuf[buf][0][0];
#pragma unroll
    for (int p = 0; p < 2; ++p) {
      int chunk = tid + 128 * p;               // 0..255
      int d = chunk >> 2, seg = chunk & 3;
      const char* vg = (const char*)(vTh + (size_t)d * N_TOK + j + seg * 8);
      async_g2l_b128(vl + (uint32_t)chunk * 16u, vg);
    }
  };

  stage(0, 0);

  for (int j = 0, it = 0; j < N_TOK; j += 32, ++it) {
    int cur = it & 1;
    if (j + 32 < N_TOK) {
      stage(j + 32, cur ^ 1);                  // prefetch next tile
      asm volatile("s_wait_asynccnt 0x4" ::: "memory");  // drain current tile only
    } else {
      asm volatile("s_wait_asynccnt 0x0" ::: "memory");
    }
    __syncthreads();                           // all waves' tile data visible

    const __bf16* kt = &kbuf[cur][0][0];       // [32][64]
    const __bf16* vt = &vbuf[cur][0][0];       // [64][32]

    // scores: S0 = Q @ K[0..15]^T, S1 = Q @ K[16..31]^T (head dim chained 2xK32)
    v8f s0 = zero8(), s1 = zero8();
    {
      v16bf b;
      b = load_b(kt, 0,  HD, 0,  lane); s0 = bmma(qa0, b, s0);
      b = load_b(kt, 0,  HD, 32, lane); s0 = bmma(qa1, b, s0);
      b = load_b(kt, 16, HD, 0,  lane); s1 = bmma(qa0, b, s1);
      b = load_b(kt, 16, HD, 32, lane); s1 = bmma(qa1, b, s1);
    }

    // online softmax (base 2; row spans the 16 lanes of a half-group)
    float alpha[8];
#pragma unroll
    for (int r = 0; r < 8; ++r) {
      float a0 = s0[r] * SCALE2;
      float a1 = s1[r] * SCALE2;
      float t  = fmaxf(a0, a1);
#pragma unroll
      for (int msk = 1; msk < 16; msk <<= 1) t = fmaxf(t, __shfl_xor(t, msk, 32));
      float mn = fmaxf(mrow[r], t);
      alpha[r] = exp2_fast(mrow[r] - mn);      // 0 on first iteration
      mrow[r]  = mn;
      a0 = exp2_fast(a0 - mn);
      a1 = exp2_fast(a1 - mn);
      s0[r] = a0; s1[r] = a1;
      float rs = a0 + a1;
#pragma unroll
      for (int msk = 1; msk < 16; msk <<= 1) rs += __shfl_xor(rs, msk, 32);
      lrow[r] = lrow[r] * alpha[r] + rs;
    }
#pragma unroll
    for (int t = 0; t < 4; ++t)
#pragma unroll
      for (int r = 0; r < 8; ++r) acc[t][r] *= alpha[r];

    // stage P (16x32 bf16) through LDS: C-layout -> row-major -> A-fragment
#pragma unroll
    for (int r = 0; r < 8; ++r) {
      pb[(hi * 8 + r) * 32 + c]      = (__bf16)s0[r];
      pb[(hi * 8 + r) * 32 + 16 + c] = (__bf16)s1[r];
    }
    asm volatile("s_wait_dscnt 0x0" ::: "memory");   // wave-local LDS RAW fence
    v16bf pa = load_a(pb, 0, 32, 0, lane);

    // acc += P(16x32) @ V(32x64), V fragments from LDS
#pragma unroll
    for (int t = 0; t < 4; ++t) {
      v16bf bv = load_b(vt, t * 16, 32, 0, lane);
      acc[t] = bmma(pa, bv, acc[t]);
    }
    __syncthreads();                           // protect K/V buffer reuse
  }

  // out[n][h*64 + d] = acc / l   (fp32 output)
#pragma unroll
  for (int t = 0; t < 4; ++t) {
    int dg = h * HD + t * 16 + c;
#pragma unroll
    for (int r = 0; r < 8; ++r) {
      int n = r0 + hi * 8 + r;
      out[(size_t)n * DMODEL + dg] = acc[t][r] / lrow[r];
    }
  }
}

// ---------------- launch ----------------

extern "C" void kernel_launch(void* const* d_in, const int* in_sizes, int n_in,
                              void* d_out, int out_size, void* d_ws, size_t ws_size,
                              hipStream_t stream) {
  const float* x  = (const float*)d_in[0];   // [2,1024,1024]
  const float* pw = (const float*)d_in[1];   // [1024,1024]
  const float* pb = (const float*)d_in[2];   // [1024]
  const float* wq = (const float*)d_in[3];   // [16,64,64]
  const float* wk = (const float*)d_in[4];   // [16,64,64]
  float* out = (float*)d_out;

  char* w = (char*)d_ws;
  __bf16* xb    = (__bf16*)w; w += (size_t)N_TOK * DMODEL * 2;     // 4 MB
  __bf16* wb    = (__bf16*)w; w += (size_t)DMODEL * DMODEL * 2;    // 2 MB
  __bf16* wqT   = (__bf16*)w; w += (size_t)NHEAD * HD * HD * 2;    // 128 KB
  __bf16* wkT   = (__bf16*)w; w += (size_t)NHEAD * HD * HD * 2;    // 128 KB
  __bf16* flat  = (__bf16*)w; w += (size_t)NHEAD * N_TOK * HD * 2; // 4 MB
  __bf16* flatT = (__bf16*)w; w += (size_t)NHEAD * HD * N_TOK * 2; // 4 MB
  __bf16* qb    = (__bf16*)w; w += (size_t)NHEAD * N_TOK * HD * 2; // 4 MB
  __bf16* kb    = (__bf16*)w; w += (size_t)NHEAD * N_TOK * HD * 2; // 4 MB

  cvt_bf16<<<dim3((N_TOK * DMODEL) / 256), 256, 0, stream>>>(x, xb, N_TOK * DMODEL);
  cvt_bf16<<<dim3((DMODEL * DMODEL) / 256), 256, 0, stream>>>(pw, wb, DMODEL * DMODEL);
  cvt_wt  <<<dim3((NHEAD * HD * HD) / 256), 256, 0, stream>>>(wq, wk, wqT, wkT);

  proj_gemm  <<<dim3(N_TOK / 64, DMODEL / 128), 128, 0, stream>>>(xb, wb, pb, flat, flatT);
  qk_gemm    <<<dim3(N_TOK / 64, NHEAD),        128, 0, stream>>>(flat, wqT, wkT, qb, kb);
  attn_kernel<<<dim3(N_TOK / 64, NHEAD),        128, 0, stream>>>(qb, kb, flatT, out);
}
